// GPTJAttention_52853867545237
// MI455X (gfx1250) — matrix operand order repaired and checked by reading, not compiled, not run on hardware
//
#include <hip/hip_runtime.h>
#include <hip/hip_bf16.h>

typedef __attribute__((ext_vector_type(8)))  __bf16 v8bf;
typedef __attribute__((ext_vector_type(16))) __bf16 v16bf;
typedef __attribute__((ext_vector_type(8)))  float  v8f;

constexpr int BB   = 2;
constexpr int SS   = 4096;
constexpr int HIDC = 4096;
constexpr int NHC  = 16;
constexpr int HDC  = 256;
constexpr int QKVW = 3 * HIDC;   // 12288

// ---------------- WMMA fragment helpers (CDNA5 16x16x32 bf16) ----------------
// A (16x32, 16-bit): lane = half*16 + m; VGPR0-3 hold K=half*8+0..7, VGPR4-7 hold K=16+half*8+0..7
__device__ __forceinline__ v16bf frag_a(const __bf16* base, int stride, int m, int k0, int lane) {
    int row = m + (lane & 15);
    int hf  = lane >> 4;
    const __bf16* p = base + (size_t)row * stride + k0 + hf * 8;
    union { v16bf f; v8bf h[2]; } u;
    u.h[0] = *(const v8bf*)p;          // K = half*8 + 0..7
    u.h[1] = *(const v8bf*)(p + 16);   // K = 16 + half*8 + 0..7
    return u.f;
}

// B (32x16, 16-bit): lane = kh*16 + n; VGPR0-3 hold K=kh*8+0..7, VGPR4-7 hold K=16+kh*8+0..7
__device__ __forceinline__ v16bf frag_b(const __bf16* base, int stride, int n, int k0, int lane) {
    int col = n + (lane & 15);
    int kh  = lane >> 4;
    const __bf16* p = base + (size_t)col * stride + k0 + kh * 8;
    union { v16bf f; v8bf h[2]; } u;
    u.h[0] = *(const v8bf*)p;
    u.h[1] = *(const v8bf*)(p + 16);
    return u.f;
}

__device__ __forceinline__ v8f wmma_bf16(v16bf a, v16bf b, v8f c) {
    return __builtin_amdgcn_wmma_f32_16x16x32_bf16(false, a, false, b, (short)0, c, false, false);
}

// pack 8 fp32 -> v8bf (lets the backend use packed converts + one 16B LDS store)
__device__ __forceinline__ v8bf pack8(float4 a, float4 b) {
    v8bf r;
    r[0] = (__bf16)a.x; r[1] = (__bf16)a.y; r[2] = (__bf16)a.z; r[3] = (__bf16)a.w;
    r[4] = (__bf16)b.x; r[5] = (__bf16)b.y; r[6] = (__bf16)b.z; r[7] = (__bf16)b.w;
    return r;
}

// ---------------- GEMM: C[M,N] = A[M,K] * B[N,K]^T ----------------
// A: fp32 or bf16 (AF32), B: fp32, C: bf16 or fp32 (OUTBF16). M%128==0, N%128==0, K%32==0.
constexpr int GBM = 128, GBN = 128, GBK = 32, GPAD = 8;

template <bool AF32, bool OUTBF16>
__global__ __launch_bounds__(256) void gemm_nt(const void* __restrict__ Av,
                                               const float* __restrict__ Bw,
                                               void* __restrict__ Cv,
                                               int M, int N, int K) {
    __shared__ __bf16 As[GBM][GBK + GPAD];
    __shared__ __bf16 Bs[GBN][GBK + GPAD];
    const int t    = threadIdx.x;
    const int lane = t & 31, wave = t >> 5;
    const int wm   = wave >> 2, wn = wave & 3;         // 2x4 wave grid, wave tile 64x32
    const int m0   = blockIdx.y * GBM, n0 = blockIdx.x * GBN;
    const int lr   = t >> 1;            // 0..127
    const int lc   = (t & 1) * 16;      // 0 or 16

    v8f zero = {0.f,0.f,0.f,0.f,0.f,0.f,0.f,0.f};
    v8f acc[4][2];
#pragma unroll
    for (int mi = 0; mi < 4; ++mi)
#pragma unroll
        for (int ni = 0; ni < 2; ++ni) acc[mi][ni] = zero;

    for (int kt = 0; kt < K; kt += GBK) {
        if constexpr (AF32) {
            const float4* ap = (const float4*)((const float*)Av + (size_t)(m0 + lr) * K + kt + lc);
            float4 x0 = ap[0], x1 = ap[1], x2 = ap[2], x3 = ap[3];
            *(v8bf*)&As[lr][lc]     = pack8(x0, x1);
            *(v8bf*)&As[lr][lc + 8] = pack8(x2, x3);
        } else {
            const __bf16* ap = (const __bf16*)Av + (size_t)(m0 + lr) * K + kt + lc;
            *(v8bf*)&As[lr][lc]     = *(const v8bf*)ap;
            *(v8bf*)&As[lr][lc + 8] = *(const v8bf*)(ap + 8);
        }
        {
            const float4* bp = (const float4*)(Bw + (size_t)(n0 + lr) * K + kt + lc);
            float4 x0 = bp[0], x1 = bp[1], x2 = bp[2], x3 = bp[3];
            *(v8bf*)&Bs[lr][lc]     = pack8(x0, x1);
            *(v8bf*)&Bs[lr][lc + 8] = pack8(x2, x3);
        }
        __syncthreads();

        v16bf af[4], bf[2];
#pragma unroll
        for (int mi = 0; mi < 4; ++mi)
            af[mi] = frag_a(&As[0][0], GBK + GPAD, wm * 64 + mi * 16, 0, lane);
#pragma unroll
        for (int ni = 0; ni < 2; ++ni)
            bf[ni] = frag_b(&Bs[0][0], GBK + GPAD, wn * 32 + ni * 16, 0, lane);
#pragma unroll
        for (int mi = 0; mi < 4; ++mi)
#pragma unroll
            for (int ni = 0; ni < 2; ++ni)
                acc[mi][ni] = wmma_bf16(af[mi], bf[ni], acc[mi][ni]);
        __syncthreads();
    }

    const int hf = lane >> 4, l15 = lane & 15;
#pragma unroll
    for (int mi = 0; mi < 4; ++mi)
#pragma unroll
        for (int ni = 0; ni < 2; ++ni)
#pragma unroll
            for (int r = 0; r < 8; ++r) {
                int row = m0 + wm * 64 + mi * 16 + hf * 8 + r;
                int col = n0 + wn * 32 + ni * 16 + l15;
                if constexpr (OUTBF16)
                    ((__bf16*)Cv)[(size_t)row * N + col] = (__bf16)acc[mi][ni][r];
                else
                    ((float*)Cv)[(size_t)row * N + col] = acc[mi][ni][r];
            }
}

// ---------------- RoPE (GPT-J interleaved), first 64 dims of q and k ----------------
__global__ __launch_bounds__(256) void rope_kernel(__bf16* __restrict__ qkv,
                                                   const int* __restrict__ pos) {
    int wid  = blockIdx.x * 8 + (threadIdx.x >> 5);   // one wave per (b,s,h)
    int lane = threadIdx.x & 31;                      // pair index i -> dims 2i,2i+1
    int h    = wid & (NHC - 1);
    int bs   = wid >> 4;                              // b*SS + s
    float p  = (float)pos[bs];
    float inv = __powf(10000.0f, -((float)(2 * lane) / 64.0f));
    float f = p * inv;
    float sn, c;
    __sincosf(f, &sn, &c);
    size_t base = (size_t)bs * QKVW + h * HDC;
#pragma unroll
    for (int part = 0; part < 2; ++part) {            // q then k
        size_t o = base + part * HIDC + 2 * lane;
        float x1 = (float)qkv[o];
        float x2 = (float)qkv[o + 1];
        qkv[o]     = (__bf16)(x1 * c - x2 * sn);
        qkv[o + 1] = (__bf16)(x2 * c + x1 * sn);
    }
}

// ---------------- Flash attention: Br=128 (8 waves x 16 rows), Bc=32, D=256 ----------------
constexpr int FBR = 128, FBC = 32, FPAD = 8;

__global__ __launch_bounds__(256) void flash_kernel(const __bf16* __restrict__ qkv,
                                                    __bf16* __restrict__ attn) {
    __shared__ __bf16 Ks[FBC][HDC + FPAD];       // 32 x 264 (keys row-major over d)
    __shared__ __bf16 Vt[HDC][FBC + FPAD];       // 256 x 40 (V transposed: [d][key])
    __shared__ __bf16 Ps[FBR][FBC + FPAD];       // probs, per-wave-private 16-row bands

    const int t = threadIdx.x, lane = t & 31, wave = t >> 5;
    const int qb = blockIdx.x, h = blockIdx.y, b = blockIdx.z;
    const int wbase = wave * 16;
    const int l15 = lane & 15, hf = lane >> 4;
    const float scale = 0.0625f;                 // 1/sqrt(256)

    // Q fragments resident in registers (rows wbase..wbase+15 of this q block)
    const __bf16* qp = qkv + ((size_t)(b * SS) + qb * FBR) * QKVW + h * HDC;
    v16bf qf[8];
#pragma unroll
    for (int kk = 0; kk < 8; ++kk)
        qf[kk] = frag_a(qp, QKVW, wbase, kk * 32, lane);

    v8f zero = {0.f,0.f,0.f,0.f,0.f,0.f,0.f,0.f};
    v8f acc[16];
#pragma unroll
    for (int nt = 0; nt < 16; ++nt) acc[nt] = zero;
    float mrow[8], lrow[8];
#pragma unroll
    for (int r = 0; r < 8; ++r) { mrow[r] = -3.0e38f; lrow[r] = 0.0f; }

    const int kbmax = qb * (FBR / FBC) + (FBR / FBC) - 1;   // causal block bound
    for (int kb = 0; kb <= kbmax; ++kb) {
        __syncthreads();   // previous iteration done with Ks/Vt
        // K tile: [key][d] bf16, 8 bf16 per thread per pass
#pragma unroll
        for (int i = 0; i < 4; ++i) {
            int key = (t >> 5) + i * 8;
            int d0  = (t & 31) * 8;
            *(v8bf*)&Ks[key][d0] =
                *(const v8bf*)(qkv + ((size_t)(b * SS) + kb * FBC + key) * QKVW + HIDC + h * HDC + d0);
        }
        // V tile transposed: Vt[d][key]
#pragma unroll
        for (int i = 0; i < 4; ++i) {
            int key = (t >> 5) + i * 8;
            int d0  = (t & 31) * 8;
            v8bf v = *(const v8bf*)(qkv + ((size_t)(b * SS) + kb * FBC + key) * QKVW + 2 * HIDC + h * HDC + d0);
#pragma unroll
            for (int j = 0; j < 8; ++j) Vt[d0 + j][key] = v[j];
        }
        __syncthreads();

        // S = Q K^T  (16 rows x 32 cols per wave)
        v8f s0 = zero, s1 = zero;
#pragma unroll
        for (int kk = 0; kk < 8; ++kk) {
            v16bf b0 = frag_b(&Ks[0][0], HDC + FPAD, 0,  kk * 32, lane);
            v16bf b1 = frag_b(&Ks[0][0], HDC + FPAD, 16, kk * 32, lane);
            s0 = wmma_bf16(qf[kk], b0, s0);
            s1 = wmma_bf16(qf[kk], b1, s1);
        }

        // online softmax: rows live in lane-halves; reduce across 16 lanes of a half
#pragma unroll
        for (int r = 0; r < 8; ++r) {
            int qg = qb * FBR + wbase + hf * 8 + r;     // global query index
            int c0 = kb * FBC + l15;
            int c1 = kb * FBC + 16 + l15;
            bool ok0 = c0 <= qg, ok1 = c1 <= qg;
            float v0 = ok0 ? s0[r] * scale : -3.0e38f;
            float v1 = ok1 ? s1[r] * scale : -3.0e38f;
            float mx = fmaxf(v0, v1);
#pragma unroll
            for (int off = 1; off < 16; off <<= 1) mx = fmaxf(mx, __shfl_xor(mx, off, 32));
            float mnew = fmaxf(mrow[r], mx);
            float p0 = ok0 ? __expf(v0 - mnew) : 0.0f;
            float p1 = ok1 ? __expf(v1 - mnew) : 0.0f;
            float sum = p0 + p1;
#pragma unroll
            for (int off = 1; off < 16; off <<= 1) sum += __shfl_xor(sum, off, 32);
            float corr = __expf(mrow[r] - mnew);
            lrow[r] = lrow[r] * corr + sum;
            mrow[r] = mnew;
#pragma unroll
            for (int nt = 0; nt < 16; ++nt) acc[nt][r] *= corr;
            int row = wbase + hf * 8 + r;
            Ps[row][l15]      = (__bf16)p0;
            Ps[row][16 + l15] = (__bf16)p1;
        }

        // O += P V  (P rows are wave-private; Vt loaded before barrier)
        v16bf pa = frag_a(&Ps[0][0], FBC + FPAD, wbase, 0, lane);
#pragma unroll
        for (int nt = 0; nt < 16; ++nt) {
            v16bf vb = frag_b(&Vt[0][0], FBC + FPAD, nt * 16, 0, lane);
            acc[nt] = wmma_bf16(pa, vb, acc[nt]);
        }
    }

    // epilogue: normalize and store bf16 attn [b, q, h*256 + d]
#pragma unroll
    for (int r = 0; r < 8; ++r) {
        float inv = 1.0f / lrow[r];
        int row = qb * FBR + wbase + hf * 8 + r;
        size_t ob = ((size_t)(b * SS) + row) * HIDC + h * HDC + l15;
#pragma unroll
        for (int nt = 0; nt < 16; ++nt)
            attn[ob + nt * 16] = (__bf16)(acc[nt][r] * inv);
    }
}

// ---------------- launch ----------------
extern "C" void kernel_launch(void* const* d_in, const int* in_sizes, int n_in,
                              void* d_out, int out_size, void* d_ws, size_t ws_size,
                              hipStream_t stream) {
    (void)in_sizes; (void)n_in; (void)out_size; (void)ws_size;
    const float* hidden = (const float*)d_in[0];
    const int*   pos    = (const int*)d_in[1];
    const float* wqkv   = (const float*)d_in[2];
    const float* wout   = (const float*)d_in[3];
    float*       out    = (float*)d_out;

    __bf16* qkv  = (__bf16*)d_ws;                                            // [B*S, 12288] bf16
    __bf16* attn = (__bf16*)((char*)d_ws + (size_t)BB * SS * QKVW * 2);      // [B*S, 4096]  bf16

    dim3 blk(256);
    // 1) qkv = hidden @ Wqkv^T   (fp32 in, bf16 out)
    gemm_nt<true, true><<<dim3(QKVW / GBN, (BB * SS) / GBM), blk, 0, stream>>>(
        hidden, wqkv, qkv, BB * SS, QKVW, HIDC);
    // 2) RoPE on q,k (first 64 dims per head)
    rope_kernel<<<(BB * SS * NHC) / 8, blk, 0, stream>>>(qkv, pos);
    // 3) causal flash attention
    flash_kernel<<<dim3(SS / FBR, NHC, BB), blk, 0, stream>>>(qkv, attn);
    // 4) out = attn @ Wout^T     (bf16 A, fp32 B, fp32 out)
    gemm_nt<false, false><<<dim3(HIDC / GBN, (BB * SS) / GBM), blk, 0, stream>>>(
        attn, wout, out, BB * SS, HIDC, HIDC);
}
